// DenseEquivariantIrrep_35253091565744
// MI455X (gfx1250) — matrix operand order repaired and checked by reading, not compiled, not run on hardware
//
#include <hip/hip_runtime.h>

// ---------------------------------------------------------------------------
// DenseEquivariantIrrep for D_32 on MI455X (gfx1250).
// The layer is linear in x: build fused W_eff (1024x1024) once, split into
// bf16 hi/lo (3xBF16 f32 emulation), then one GEMM:
//     out(65536x1024) = x(65536x1024) @ W_eff^T + bias
// with v_wmma_f32_16x16x32_bf16. W tiles staged via
// global_load_async_to_lds_b128 (ASYNCcnt); x split f32->bf16 hi/lo using
// truncation + v_perm_b32 packing (residual captured exactly in lo).
// ---------------------------------------------------------------------------

typedef __attribute__((ext_vector_type(16))) __bf16 v16bf;
typedef __attribute__((ext_vector_type(8)))  __bf16 v8bf;
typedef __attribute__((ext_vector_type(8)))  float  v8f;

static __device__ __forceinline__ __bf16 f2bf(float f) {
    unsigned u = __builtin_bit_cast(unsigned, f);
    unsigned r = u + 0x7FFFu + ((u >> 16) & 1u);            // RNE (setup kernel only)
    unsigned short hv = (unsigned short)(r >> 16);
    return __builtin_bit_cast(__bf16, hv);
}
static __device__ __forceinline__ float bf2f(__bf16 b) {
    unsigned short s = __builtin_bit_cast(unsigned short, b);
    unsigned u = ((unsigned)s) << 16;
    return __builtin_bit_cast(float, u);
}
// one v_perm_b32: result = { b[31:16], a[31:16] }  (packed v2bf16: elem0=a, elem1=b)
static __device__ __forceinline__ unsigned pk_trunc(float a, float b) {
    return __builtin_amdgcn_perm(__builtin_bit_cast(unsigned, b),
                                 __builtin_bit_cast(unsigned, a), 0x07060302u);
}
static __device__ __forceinline__ v16bf cat8(v8bf a, v8bf b) {
    return __builtin_shufflevector(a, b, 0,1,2,3,4,5,6,7,8,9,10,11,12,13,14,15);
}

// ---------------------------------------------------------------------------
// Setup: one block per (f,c). W_fc[g'][g] = sum_t inv[t][g'] * M[t][g] where M
// encodes the per-irrep block products with kh = kernel_row @ fwd.
// ---------------------------------------------------------------------------
__global__ __launch_bounds__(256)
void build_weff_kernel(const float* __restrict__ kparams,   // (16,16,32)
                       const int*   __restrict__ kidx,      // (32)
                       const float* __restrict__ fwd,       // (64,64) [g][t]
                       const float* __restrict__ inv,       // (64,64) [t][g']
                       __bf16* __restrict__ Whi,            // (1024,1024)
                       __bf16* __restrict__ Wlo)
{
    __shared__ float kh[64];
    __shared__ float Mm[64][65];

    const int fc  = blockIdx.x;
    const int f   = fc >> 4;
    const int c   = fc & 15;
    const int tid = threadIdx.x;

    if (tid < 64) {
        float s = 0.f;
        const float* kp = kparams + (size_t)(f * 16 + c) * 32;
        #pragma unroll 8
        for (int i = 0; i < 32; ++i)
            s += kp[i] * fwd[kidx[i] * 64 + tid];
        kh[tid] = s;
    }
    __syncthreads();

    #pragma unroll 1
    for (int e = 0; e < 16; ++e) {
        int id = tid * 16 + e;            // 0..4095
        int t = id >> 6, g = id & 63;
        float m;
        if (t < 4) {
            m = kh[t] * fwd[g * 64 + t];
        } else {
            int n = (t - 4) >> 2, r = (t - 4) & 3;
            int i = r >> 1, k = r & 1;
            int base = 4 + 4 * n;
            m = kh[base + k]     * fwd[g * 64 + base + 2 * i]
              + kh[base + 2 + k] * fwd[g * 64 + base + 2 * i + 1];
        }
        Mm[t][g] = m;
    }
    __syncthreads();

    #pragma unroll 1
    for (int e = 0; e < 16; ++e) {
        int id = tid * 16 + e;
        int gp = id >> 6, g = id & 63;
        float acc = 0.f;
        #pragma unroll 8
        for (int t = 0; t < 64; ++t)
            acc += inv[t * 64 + gp] * Mm[t][g];
        __bf16 h = f2bf(acc);
        __bf16 l = f2bf(acc - bf2f(h));
        size_t o = (size_t)(f * 64 + gp) * 1024 + (size_t)(c * 64 + g);
        Whi[o] = h;
        Wlo[o] = l;
    }
}

// ---------------------------------------------------------------------------
// Main GEMM: M=65536, N=1024, K=1024, block tile 128x128, K-step 32,
// 8 wave32 per block (4x2), wave tile 32x64, 3xBF16 split.
// ---------------------------------------------------------------------------
#define BM 128
#define BN 128
#define BK 32
#define LDT 40   // LDS row stride in bf16 (80 B, 16 B-aligned)

__global__ __launch_bounds__(256)
void irrep_gemm_kernel(const float*  __restrict__ x,     // (65536,1024)
                       const __bf16* __restrict__ Whi,   // (1024,1024) [n][k]
                       const __bf16* __restrict__ Wlo,
                       const float*  __restrict__ bias,  // (16)
                       float*        __restrict__ out)   // (65536,1024)
{
    __shared__ __bf16 sAhi[BM * LDT];
    __shared__ __bf16 sAlo[BM * LDT];
    __shared__ __bf16 sBhi[BN * LDT];
    __shared__ __bf16 sBlo[BN * LDT];

    const int tid  = threadIdx.x;
    const int lane = tid & 31;
    const int wid  = tid >> 5;
    const int bm   = blockIdx.y * BM;
    const int bn   = blockIdx.x * BN;

    const int wm = (wid >> 1) * 32;
    const int wn = (wid & 1) * 64;
    const int lr = lane & 15;
    const int lh = lane >> 4;

    v8f acc[2][4];
    #pragma unroll
    for (int mi = 0; mi < 2; ++mi)
        #pragma unroll
        for (int ni = 0; ni < 4; ++ni)
            acc[mi][ni] = (v8f)0.f;

    const int lrow = tid >> 1;            // 0..127
    const int lcol = (tid & 1) * 16;      // 0 or 16
    const int K = 1024;

    // raw LDS byte offsets for async B staging (low 32 bits of flat address)
    const unsigned ldsBhi = (unsigned)(size_t)(void*)&sBhi[lrow * LDT + lcol];
    const unsigned ldsBlo = (unsigned)(size_t)(void*)&sBlo[lrow * LDT + lcol];
    const unsigned rowByteOff = (unsigned)((bn + lrow) * (K * 2));  // W row start

    for (int k0 = 0; k0 < K; k0 += BK) {
        // ---- stage B tiles: async global->LDS copies (ASYNCcnt) ----
        {
            unsigned voff = rowByteOff + (unsigned)((k0 + lcol) * 2);
            asm volatile(
                "global_load_async_to_lds_b128 %0, %2, %3\n\t"
                "global_load_async_to_lds_b128 %0, %2, %3 offset:16\n\t"
                "global_load_async_to_lds_b128 %1, %2, %4\n\t"
                "global_load_async_to_lds_b128 %1, %2, %4 offset:16"
                :
                : "v"(ldsBhi), "v"(ldsBlo), "v"(voff), "s"(Whi), "s"(Wlo)
                : "memory");
        }
        // ---- stage A: load 16 f32 of x, truncation-split into bf16 hi/lo ----
        {
            const float4* xp =
                (const float4*)(x + (size_t)(bm + lrow) * K + k0 + lcol);
            float4 f0 = xp[0], f1 = xp[1], f2 = xp[2], f3 = xp[3];
            float v[16] = { f0.x,f0.y,f0.z,f0.w, f1.x,f1.y,f1.z,f1.w,
                            f2.x,f2.y,f2.z,f2.w, f3.x,f3.y,f3.z,f3.w };
            unsigned hp[8], lp[8];
            #pragma unroll
            for (int j = 0; j < 8; ++j) {
                float a = v[2*j], b = v[2*j+1];
                hp[j] = pk_trunc(a, b);                    // 1x v_perm_b32
                unsigned ua = __builtin_bit_cast(unsigned, a) & 0xFFFF0000u;
                unsigned ub = __builtin_bit_cast(unsigned, b) & 0xFFFF0000u;
                float ra = a - __builtin_bit_cast(float, ua);   // exact residual
                float rb = b - __builtin_bit_cast(float, ub);
                lp[j] = pk_trunc(ra, rb);                  // 1x v_perm_b32
            }
            uint4* da = (uint4*)&sAhi[lrow * LDT + lcol];
            da[0] = uint4{hp[0], hp[1], hp[2], hp[3]};
            da[1] = uint4{hp[4], hp[5], hp[6], hp[7]};
            uint4* dl = (uint4*)&sAlo[lrow * LDT + lcol];
            dl[0] = uint4{lp[0], lp[1], lp[2], lp[3]};
            dl[1] = uint4{lp[4], lp[5], lp[6], lp[7]};
        }
        // prefetch next K-panel of x (global_prefetch_b8)
        if (k0 + BK < K)
            __builtin_prefetch(x + (size_t)(bm + lrow) * K + (k0 + BK) + lcol, 0, 0);

        // wait for this wave's async B copies, then workgroup barrier
        asm volatile("s_wait_asynccnt 0x0" ::: "memory");
        __syncthreads();

        // ---- fragments per documented CDNA5 WMMA layouts ----
        v16bf Ahi[2], Alo[2], Bhi[4], Blo[4];
        #pragma unroll
        for (int mi = 0; mi < 2; ++mi) {
            const __bf16* p = &sAhi[(wm + mi * 16 + lr) * LDT];
            Ahi[mi] = cat8(*(const v8bf*)(p + 8 * lh), *(const v8bf*)(p + 16 + 8 * lh));
            const __bf16* q = &sAlo[(wm + mi * 16 + lr) * LDT];
            Alo[mi] = cat8(*(const v8bf*)(q + 8 * lh), *(const v8bf*)(q + 16 + 8 * lh));
        }
        #pragma unroll
        for (int ni = 0; ni < 4; ++ni) {
            const __bf16* p = &sBhi[(wn + ni * 16 + lr) * LDT];
            Bhi[ni] = cat8(*(const v8bf*)(p + 8 * lh), *(const v8bf*)(p + 16 + 8 * lh));
            const __bf16* q = &sBlo[(wn + ni * 16 + lr) * LDT];
            Blo[ni] = cat8(*(const v8bf*)(q + 8 * lh), *(const v8bf*)(q + 16 + 8 * lh));
        }

        #pragma unroll
        for (int mi = 0; mi < 2; ++mi)
            #pragma unroll
            for (int ni = 0; ni < 4; ++ni) {
                acc[mi][ni] = __builtin_amdgcn_wmma_f32_16x16x32_bf16(
                    false, Ahi[mi], false, Bhi[ni], (short)0, acc[mi][ni], false, false);
                acc[mi][ni] = __builtin_amdgcn_wmma_f32_16x16x32_bf16(
                    false, Alo[mi], false, Bhi[ni], (short)0, acc[mi][ni], false, false);
                acc[mi][ni] = __builtin_amdgcn_wmma_f32_16x16x32_bf16(
                    false, Ahi[mi], false, Blo[ni], (short)0, acc[mi][ni], false, false);
            }
        __syncthreads();
    }

    // ---- epilogue: C/D layout -> global, + bias ----
    #pragma unroll
    for (int mi = 0; mi < 2; ++mi)
        #pragma unroll
        for (int ni = 0; ni < 4; ++ni) {
            int n = bn + wn + ni * 16 + lr;
            float bv = bias[n >> 6];
            int mbase = bm + wm + mi * 16 + 8 * lh;
            #pragma unroll
            for (int r = 0; r < 8; ++r)
                out[(size_t)(mbase + r) * 1024 + n] = acc[mi][ni][r] + bv;
        }
}

// ---------------------------------------------------------------------------
extern "C" void kernel_launch(void* const* d_in, const int* in_sizes, int n_in,
                              void* d_out, int out_size, void* d_ws, size_t ws_size,
                              hipStream_t stream) {
    const float* x       = (const float*)d_in[0];   // (65536,16,64)
    const float* kparams = (const float*)d_in[1];   // (16,16,32)
    const float* bias    = (const float*)d_in[2];   // (16)
    const int*   kidx    = (const int*)d_in[3];     // (32)
    const float* fwd     = (const float*)d_in[4];   // (64,64)
    const float* inv     = (const float*)d_in[5];   // (64,64)
    float*       out     = (float*)d_out;           // (65536,16,64)

    __bf16* Whi = (__bf16*)d_ws;                    // 2 MB
    __bf16* Wlo = Whi + (size_t)1024 * 1024;        // 2 MB

    build_weff_kernel<<<256, 256, 0, stream>>>(kparams, kidx, fwd, inv, Whi, Wlo);

    dim3 grid(1024 / BN, 65536 / BM);               // (8, 512)
    irrep_gemm_kernel<<<grid, 256, 0, stream>>>(x, Whi, Wlo, bias, out);
}
